// XAttn_26456998543684
// MI455X (gfx1250) — matrix operand order repaired and checked by reading, not compiled, not run on hardware
//
#include <hip/hip_runtime.h>
#include <hip/hip_bf16.h>

// ---------- types ----------
typedef __bf16 bf16;
typedef __attribute__((ext_vector_type(16))) __bf16 v16bf;
typedef __attribute__((ext_vector_type(8)))  float  v8f;

union FragA { v16bf v; bf16 h[16]; };
union FragC { v8f  v; float f[8]; };

#define DIM    1024
#define NHEADS 16
#define HD     64
#define BATCH  2
#define SEQ    2048
#define ROWS   (BATCH * SEQ)   // 4096

// ---------- elementwise f32 -> bf16 ----------
__global__ void cvt_f32_bf16(const float* __restrict__ in, bf16* __restrict__ out, int n) {
    int i = blockIdx.x * blockDim.x + threadIdx.x;
    int stride = gridDim.x * blockDim.x;
    for (; i < n; i += stride) out[i] = (bf16)in[i];
}

// ---------- C[4096,1024] = A[4096,1024](bf16) x W[1024,1024]^T(bf16) + bias ----------
// Row-major W[n][k] is already the K-contiguous layout the B-fragment wants.
// 128x128 block tile, 8 waves (4 row-groups x 2 col-groups), each wave 32x64.
// Double-buffered LDS; next tile's global loads issued before current tile's WMMAs.
// MODE 0: f32 row-major; MODE 1: bf16 row-major; MODE 2: bf16 vT[(b*1024+col)][key].
template<int MODE>
__global__ __launch_bounds__(256) void gemm_bias_kernel(
    const bf16* __restrict__ A, const bf16* __restrict__ W,
    const float* __restrict__ bias, float* __restrict__ Cf, bf16* __restrict__ Cb)
{
    __shared__ bf16 As[2][128][48];
    __shared__ bf16 Ws[2][128][48];

    const int tid  = threadIdx.x;
    const int wave = tid >> 5;
    const int lane = tid & 31;
    const int half = lane >> 4;
    const int low  = lane & 15;
    const int rowBase = blockIdx.y * 128;
    const int colBase = blockIdx.x * 128;
    const int wr = (wave & 3) * 32;
    const int wc = (wave >> 2) * 64;

    // fixed staging assignment: thread -> rows {r0, r0+64}, bf16 columns ss..ss+7
    const int r0 = tid >> 2;            // 0..63
    const int ss = (tid & 3) * 8;       // 0,8,16,24
    const bf16* aPtr = A + (size_t)(rowBase + r0) * DIM + ss;
    const bf16* wPtr = W + (size_t)(colBase + r0) * DIM + ss;
    const size_t rstep = (size_t)64 * DIM;

    FragC acc[2][4];
    #pragma unroll
    for (int mi = 0; mi < 2; mi++)
        #pragma unroll
        for (int ni = 0; ni < 4; ni++)
            #pragma unroll
            for (int e = 0; e < 8; e++) acc[mi][ni].f[e] = 0.0f;

    // prologue: stage tile 0
    uint4 a0 = *(const uint4*)(aPtr);
    uint4 a1 = *(const uint4*)(aPtr + rstep);
    uint4 w0 = *(const uint4*)(wPtr);
    uint4 w1 = *(const uint4*)(wPtr + rstep);
    *(uint4*)&As[0][r0     ][ss] = a0;
    *(uint4*)&As[0][r0 + 64][ss] = a1;
    *(uint4*)&Ws[0][r0     ][ss] = w0;
    *(uint4*)&Ws[0][r0 + 64][ss] = w1;
    __syncthreads();

    const int NT = DIM / 32;
    for (int kt = 0; kt < NT; ++kt) {
        const int cur = kt & 1;
        // issue next tile's global loads early (latency hidden under WMMAs)
        if (kt + 1 < NT) {
            const bf16* ap = aPtr + (kt + 1) * 32;
            const bf16* wp = wPtr + (kt + 1) * 32;
            a0 = *(const uint4*)(ap);
            a1 = *(const uint4*)(ap + rstep);
            w0 = *(const uint4*)(wp);
            w1 = *(const uint4*)(wp + rstep);
        }

        // A fragment (16x32): two contiguous 16B runs per lane (2x ds_load_b128)
        FragA afrag[2], bfrag[4];
        #pragma unroll
        for (int mi = 0; mi < 2; mi++)
            #pragma unroll
            for (int e = 0; e < 16; e++) {
                int kidx = ((e >> 3) << 4) + (half << 3) + (e & 7);
                afrag[mi].h[e] = As[cur][wr + mi * 16 + low][kidx];
            }
        // B fragment (32x16): two contiguous 16B runs per lane (2x ds_load_b128)
        #pragma unroll
        for (int ni = 0; ni < 4; ni++)
            #pragma unroll
            for (int e = 0; e < 16; e++)
                bfrag[ni].h[e] = Ws[cur][wc + ni * 16 + low][((e >> 3) << 4) + (half << 3) + (e & 7)];

        #pragma unroll
        for (int mi = 0; mi < 2; mi++)
            #pragma unroll
            for (int ni = 0; ni < 4; ni++)
                acc[mi][ni].v = __builtin_amdgcn_wmma_f32_16x16x32_bf16(
                    false, afrag[mi].v, false, bfrag[ni].v,
                    (short)0, acc[mi][ni].v, false, false);

        // stash next tile into the other buffer
        if (kt + 1 < NT) {
            const int nxt = cur ^ 1;
            *(uint4*)&As[nxt][r0     ][ss] = a0;
            *(uint4*)&As[nxt][r0 + 64][ss] = a1;
            *(uint4*)&Ws[nxt][r0     ][ss] = w0;
            *(uint4*)&Ws[nxt][r0 + 64][ss] = w1;
        }
        __syncthreads();
    }

    // epilogue: C layout VGPR r -> M = r + half*8, N = low
    #pragma unroll
    for (int ni = 0; ni < 4; ni++) {
        const int gcol = colBase + wc + ni * 16 + low;
        const float bia = bias[gcol];
        #pragma unroll
        for (int mi = 0; mi < 2; mi++)
            #pragma unroll
            for (int r = 0; r < 8; r++) {
                int grow = rowBase + wr + mi * 16 + half * 8 + r;
                float v = acc[mi][ni].f[r] + bia;
                if (MODE == 0) {
                    Cf[(size_t)grow * DIM + gcol] = v;
                } else if (MODE == 1) {
                    Cb[(size_t)grow * DIM + gcol] = (bf16)v;
                } else {
                    int bb  = grow >> 11;          // batch
                    int key = grow & (SEQ - 1);    // sequence position
                    Cb[(size_t)(bb * DIM + gcol) * SEQ + key] = (bf16)v;
                }
            }
    }
}

// ---------- flash attention: 64 q-rows per block, 4 waves x 16 rows ----------
// V comes pre-transposed: Vt[(b*1024 + h*64 + d)][key]. Double-buffered K/V tiles.
__global__ __launch_bounds__(128) void attn_kernel(
    const bf16* __restrict__ Q, const bf16* __restrict__ K,
    const bf16* __restrict__ Vt, bf16* __restrict__ O)
{
    __shared__ bf16 Qs[64][72];       // [q-row][d]
    __shared__ bf16 Ks[2][64][72];    // [key][d]
    __shared__ bf16 Vts[2][64][72];   // [d][key]
    __shared__ bf16 Ps[4][16][72];    // per-wave P staging (C-layout -> A-layout)

    const int tid  = threadIdx.x;
    const int wave = tid >> 5;
    const int lane = tid & 31;
    const int half = lane >> 4;
    const int low  = lane & 15;
    const int bh = blockIdx.y;
    const int b  = bh >> 4;
    const int h  = bh & 15;
    const int qbase = blockIdx.x * 64;
    const size_t rowOff = (size_t)b * SEQ;
    const int colOff = h * HD;
    const int vrow0  = b * DIM + h * HD;
    const float scale = 0.125f;       // 1/sqrt(64)

    // fixed staging assignment: thread -> rows {r0, r0+16, r0+32, r0+48}, columns ss..ss+7
    const int r0 = tid >> 3;          // 0..15
    const int ss = (tid & 7) * 8;     // 0..56
    const bf16* kPtr = K  + (rowOff + r0) * DIM + colOff + ss;
    const bf16* vPtr = Vt + (size_t)(vrow0 + r0) * SEQ + ss;

    // stage Q tile 64x64 (read once)
    #pragma unroll
    for (int j = 0; j < 4; ++j)
        *(uint4*)&Qs[r0 + 16 * j][ss] =
            *(const uint4*)&Q[(rowOff + qbase + r0 + 16 * j) * DIM + colOff + ss];

    // prologue: stage K/V tile 0
    uint4 kr[4], vr[4];
    #pragma unroll
    for (int j = 0; j < 4; ++j) kr[j] = *(const uint4*)(kPtr + (size_t)(16 * j) * DIM);
    #pragma unroll
    for (int j = 0; j < 4; ++j) vr[j] = *(const uint4*)(vPtr + (size_t)(16 * j) * SEQ);
    #pragma unroll
    for (int j = 0; j < 4; ++j) *(uint4*)&Ks[0][r0 + 16 * j][ss] = kr[j];
    #pragma unroll
    for (int j = 0; j < 4; ++j) *(uint4*)&Vts[0][r0 + 16 * j][ss] = vr[j];
    __syncthreads();

    // Q A-fragments for the two K=32 steps over head_dim (contiguous 16B runs)
    FragA qf[2];
    #pragma unroll
    for (int ds = 0; ds < 2; ds++)
        #pragma unroll
        for (int e = 0; e < 16; e++) {
            int d = ds * 32 + ((e >> 3) << 4) + (half << 3) + (e & 7);
            qf[ds].h[e] = Qs[wave * 16 + low][d];
        }

    float mrow[8], lrow[8];
    FragC o[4];
    #pragma unroll
    for (int r = 0; r < 8; r++) { mrow[r] = -1e30f; lrow[r] = 0.0f; }
    #pragma unroll
    for (int dt = 0; dt < 4; dt++)
        #pragma unroll
        for (int e = 0; e < 8; e++) o[dt].f[e] = 0.0f;

    const int NT = SEQ / 64;
    for (int kt = 0; kt < NT; ++kt) {
        const int cur = kt & 1;
        // issue next K/V tile loads early
        if (kt + 1 < NT) {
            const bf16* kp = kPtr + (size_t)(kt + 1) * 64 * DIM;
            const bf16* vp = vPtr + (kt + 1) * 64;
            #pragma unroll
            for (int j = 0; j < 4; ++j) kr[j] = *(const uint4*)(kp + (size_t)(16 * j) * DIM);
            #pragma unroll
            for (int j = 0; j < 4; ++j) vr[j] = *(const uint4*)(vp + (size_t)(16 * j) * SEQ);
        }

        // S = Q K^T  (16 q-rows x 64 keys)
        FragC sc[4];
        #pragma unroll
        for (int nt = 0; nt < 4; nt++)
            #pragma unroll
            for (int e = 0; e < 8; e++) sc[nt].f[e] = 0.0f;

        #pragma unroll
        for (int ds = 0; ds < 2; ds++) {
            FragA bk[4];
            #pragma unroll
            for (int nt = 0; nt < 4; nt++)
                #pragma unroll
                for (int e = 0; e < 16; e++)
                    bk[nt].h[e] = Ks[cur][nt * 16 + low][ds * 32 + ((e >> 3) << 4) + (half << 3) + (e & 7)];
            #pragma unroll
            for (int nt = 0; nt < 4; nt++)
                sc[nt].v = __builtin_amdgcn_wmma_f32_16x16x32_bf16(
                    false, qf[ds].v, false, bk[nt].v, (short)0, sc[nt].v, false, false);
        }

        // online softmax; row state for q-row (r + half*8) lives in the 16 lanes of a half
        float rmax[8];
        #pragma unroll
        for (int r = 0; r < 8; r++) {
            float mx = -1e30f;
            #pragma unroll
            for (int nt = 0; nt < 4; nt++) {
                sc[nt].f[r] *= scale;
                mx = fmaxf(mx, sc[nt].f[r]);
            }
            rmax[r] = mx;
        }
        #pragma unroll
        for (int mask = 1; mask <= 8; mask <<= 1)
            #pragma unroll
            for (int r = 0; r < 8; r++)
                rmax[r] = fmaxf(rmax[r], __shfl_xor(rmax[r], mask, 32));

        float alpha[8], rsum[8];
        #pragma unroll
        for (int r = 0; r < 8; r++) {
            float nm = fmaxf(mrow[r], rmax[r]);
            alpha[r] = __expf(mrow[r] - nm);
            mrow[r] = nm;
            float s = 0.0f;
            #pragma unroll
            for (int nt = 0; nt < 4; nt++) {
                float p = __expf(sc[nt].f[r] - nm);
                sc[nt].f[r] = p;
                s += p;
            }
            rsum[r] = s;
        }
        #pragma unroll
        for (int mask = 1; mask <= 8; mask <<= 1)
            #pragma unroll
            for (int r = 0; r < 8; r++)
                rsum[r] += __shfl_xor(rsum[r], mask, 32);
        #pragma unroll
        for (int r = 0; r < 8; r++) lrow[r] = lrow[r] * alpha[r] + rsum[r];
        #pragma unroll
        for (int dt = 0; dt < 4; dt++)
            #pragma unroll
            for (int r = 0; r < 8; r++) o[dt].f[r] *= alpha[r];

        // stage P (C-layout) into per-wave LDS, reload as A-layout (same-wave DS in-order)
        #pragma unroll
        for (int nt = 0; nt < 4; nt++)
            #pragma unroll
            for (int r = 0; r < 8; r++)
                Ps[wave][half * 8 + r][nt * 16 + low] = (bf16)sc[nt].f[r];

        #pragma unroll
        for (int ks = 0; ks < 2; ks++) {
            FragA pa;
            #pragma unroll
            for (int e = 0; e < 16; e++) {
                int key = ks * 32 + ((e >> 3) << 4) + (half << 3) + (e & 7);
                pa.h[e] = Ps[wave][low][key];
            }
            FragA bvf[4];
            #pragma unroll
            for (int dt = 0; dt < 4; dt++)
                #pragma unroll
                for (int e = 0; e < 16; e++)
                    bvf[dt].h[e] = Vts[cur][dt * 16 + low][ks * 32 + ((e >> 3) << 4) + (half << 3) + (e & 7)];
            #pragma unroll
            for (int dt = 0; dt < 4; dt++)
                o[dt].v = __builtin_amdgcn_wmma_f32_16x16x32_bf16(
                    false, pa.v, false, bvf[dt].v, (short)0, o[dt].v, false, false);
        }

        // stash next tile into the other buffer
        if (kt + 1 < NT) {
            const int nxt = cur ^ 1;
            #pragma unroll
            for (int j = 0; j < 4; ++j) *(uint4*)&Ks[nxt][r0 + 16 * j][ss] = kr[j];
            #pragma unroll
            for (int j = 0; j < 4; ++j) *(uint4*)&Vts[nxt][r0 + 16 * j][ss] = vr[j];
        }
        __syncthreads();
    }

    // normalize and write merged-head layout [b, n, h*64 + d]
    #pragma unroll
    for (int dt = 0; dt < 4; dt++)
        #pragma unroll
        for (int r = 0; r < 8; r++) {
            int grow = qbase + wave * 16 + half * 8 + r;
            float val = o[dt].f[r] / lrow[r];
            O[(rowOff + grow) * DIM + colOff + dt * 16 + low] = (bf16)val;
        }
}

// ---------- launch ----------
extern "C" void kernel_launch(void* const* d_in, const int* in_sizes, int n_in,
                              void* d_out, int out_size, void* d_ws, size_t ws_size,
                              hipStream_t stream) {
    const float* x1 = (const float*)d_in[0];
    const float* x2 = (const float*)d_in[1];
    const float* Wq = (const float*)d_in[2];
    const float* bq = (const float*)d_in[3];
    const float* Wk = (const float*)d_in[4];
    const float* bk = (const float*)d_in[5];
    const float* Wv = (const float*)d_in[6];
    const float* bv = (const float*)d_in[7];
    const float* Wo = (const float*)d_in[8];
    const float* bo = (const float*)d_in[9];
    float* out = (float*)d_out;

    const size_t SZ_X = (size_t)ROWS * DIM;   // 4M elements
    const size_t SZ_W = (size_t)DIM * DIM;    // 1M elements

    bf16* ws  = (bf16*)d_ws;
    bf16* x1b = ws;               size_t off = SZ_X;
    bf16* x2b = ws + off;         off += SZ_X;
    bf16* Wqb = ws + off;         off += SZ_W;
    bf16* Wkb = ws + off;         off += SZ_W;
    bf16* Wvb = ws + off;         off += SZ_W;
    bf16* Wob = ws + off;         off += SZ_W;
    bf16* qb  = ws + off;         off += SZ_X;
    bf16* kb  = ws + off;         off += SZ_X;
    bf16* vTb = ws + off;         off += SZ_X;  // [b*1024 + h*64 + d][key]
    bf16* ob  = ws + off;         off += SZ_X;

    // 1) convert activations + weights (weights stay row-major [n][k])
    cvt_f32_bf16<<<2048, 256, 0, stream>>>(x1, x1b, (int)SZ_X);
    cvt_f32_bf16<<<2048, 256, 0, stream>>>(x2, x2b, (int)SZ_X);
    cvt_f32_bf16<<<1024, 256, 0, stream>>>(Wq, Wqb, (int)SZ_W);
    cvt_f32_bf16<<<1024, 256, 0, stream>>>(Wk, Wkb, (int)SZ_W);
    cvt_f32_bf16<<<1024, 256, 0, stream>>>(Wv, Wvb, (int)SZ_W);
    cvt_f32_bf16<<<1024, 256, 0, stream>>>(Wo, Wob, (int)SZ_W);

    // 2) projections (bf16 out; V transposed per (b,h))
    dim3 gg(DIM / 128, ROWS / 128);
    gemm_bias_kernel<1><<<gg, 256, 0, stream>>>(x1b, Wqb, bq, nullptr, qb);
    gemm_bias_kernel<1><<<gg, 256, 0, stream>>>(x2b, Wkb, bk, nullptr, kb);
    gemm_bias_kernel<2><<<gg, 256, 0, stream>>>(x2b, Wvb, bv, nullptr, vTb);

    // 3) attention
    dim3 ag(SEQ / 64, BATCH * NHEADS);
    attn_kernel<<<ag, 128, 0, stream>>>(qb, kb, vTb, ob);

    // 4) output projection (fp32 out)
    gemm_bias_kernel<0><<<gg, 256, 0, stream>>>(ob, Wob, bo, out, nullptr);
}
